// GCN_12120397709776
// MI455X (gfx1250) — matrix operand order repaired and checked by reading, not compiled, not run on hardware
//
#include <hip/hip_runtime.h>
#include <stddef.h>

// ---------------------------------------------------------------------------
// Types for WMMA
// ---------------------------------------------------------------------------
typedef __bf16           bf16x16 __attribute__((ext_vector_type(16)));
typedef unsigned short   u16x16  __attribute__((ext_vector_type(16)));
typedef unsigned short   u16x8   __attribute__((ext_vector_type(8)));
typedef unsigned short   u16x4   __attribute__((ext_vector_type(4)));
typedef float            floatx8 __attribute__((ext_vector_type(8)));

__device__ __forceinline__ unsigned short f32_to_bf16(float f) {
    unsigned int u = __builtin_bit_cast(unsigned int, f);
    unsigned int r = u + 0x7fffu + ((u >> 16) & 1u);   // round-to-nearest-even
    return (unsigned short)(r >> 16);
}

// ---------------------------------------------------------------------------
// Degree / normalization kernels
// ---------------------------------------------------------------------------
__global__ void fill_ones(float* __restrict__ p, int n) {
    int i = blockIdx.x * blockDim.x + threadIdx.x;
    if (i < n) p[i] = 1.0f;            // self-loop contribution to degree
}

__global__ void deg_scatter(const int* __restrict__ dst, float* __restrict__ deg, int E) {
    int e = blockIdx.x * blockDim.x + threadIdx.x;
    if (e < E) unsafeAtomicAdd(&deg[dst[e]], 1.0f);
}

__global__ void deg_rsqrt(float* __restrict__ p, int n) {
    int i = blockIdx.x * blockDim.x + threadIdx.x;
    if (i < n) p[i] = rsqrtf(p[i]);    // deg >= 1 always (self-loops)
}

// ---------------------------------------------------------------------------
// fp32 -> bf16 bulk conversion (activation prepass), 4 elements / thread
// ---------------------------------------------------------------------------
__global__ void to_bf16(const float* __restrict__ in, unsigned short* __restrict__ out,
                        int n4) {
    int idx = blockIdx.x * blockDim.x + threadIdx.x;
    if (idx >= n4) return;
    const float4 v = *(const float4*)(in + (size_t)idx * 4);
    u16x4 o;
    o[0] = f32_to_bf16(v.x); o[1] = f32_to_bf16(v.y);
    o[2] = f32_to_bf16(v.z); o[3] = f32_to_bf16(v.w);
    *(u16x4*)(out + (size_t)idx * 4) = o;
}

// ---------------------------------------------------------------------------
// Pre-swizzle W (128x128 fp32, row-major W[k][n]) into the CDNA5 B-fragment
// layout for V_WMMA_F32_16X16X32_BF16 (ISA 7.12.2):
// Wf[(((kt*8+nt)*32)+lane)*16 + e] = bf16( W[(kt*32 + (lane<16?0:8) + (e&7) + ((e&8)<<1))*128
//                                           + nt*16 + (lane&15)] )
// ---------------------------------------------------------------------------
__global__ void swizzle_w(const float* __restrict__ W, unsigned short* __restrict__ Wf) {
    int idx  = blockIdx.x * blockDim.x + threadIdx.x;   // 0..16383
    int e    = idx & 15;
    int lane = (idx >> 4) & 31;
    int nt   = (idx >> 9) & 7;
    int kt   = (idx >> 12) & 3;
    int n    = nt * 16 + (lane & 15);
    int k    = kt * 32 + ((lane < 16) ? 0 : 8) + (e & 7) + ((e & 8) << 1);
    Wf[idx]  = f32_to_bf16(W[k * 128 + n]);
}

// ---------------------------------------------------------------------------
// GEMM: H[N,128](f32) = Xb[N,128](bf16) @ Wf(bf16, pre-swizzled), fp32 acc.
// One wave per 16-row tile; 4 K-chunks x 8 N-tiles = 32 WMMAs.
// A fragment = two contiguous 16B bf16 loads per lane, no conversion VALU.
// ---------------------------------------------------------------------------
__global__ void __launch_bounds__(256)
gemm_wmma_bf16(const unsigned short* __restrict__ Xb,
               const unsigned short* __restrict__ Wf,
               float* __restrict__ H, int ntiles)
{
    const int wave = threadIdx.x >> 5;
    const int lane = threadIdx.x & 31;
    const int tile = blockIdx.x * 8 + wave;
    if (tile >= ntiles) return;                 // wave-uniform: EXEC stays all-1s

    const int m0  = tile << 4;
    const int row = m0 + (lane & 15);
    const int ko  = (lane >> 4) << 3;           // 0 (lanes 0-15) or 8 (16-31)

    floatx8 acc[8];
#pragma unroll
    for (int i = 0; i < 8; ++i) acc[i] = (floatx8)0.0f;

    const unsigned short* xrow = Xb + (size_t)row * 128;

#pragma unroll
    for (int kt = 0; kt < 4; ++kt) {
        const int kk = kt * 32;
        // A fragment (16x32 bf16): elems 0-7 <-> K=kk+ko+0..7,
        //                          elems 8-15 <-> K=kk+16+ko+0..7
        u16x8 alo = *(const u16x8*)(xrow + kk + ko);
        u16x8 ahi = *(const u16x8*)(xrow + kk + 16 + ko);
        u16x16 au = __builtin_shufflevector(alo, ahi,
                        0, 1, 2, 3, 4, 5, 6, 7, 8, 9, 10, 11, 12, 13, 14, 15);
        bf16x16 a = __builtin_bit_cast(bf16x16, au);

#pragma unroll
        for (int nt = 0; nt < 8; ++nt) {
            const u16x16* bp =
                (const u16x16*)(Wf + ((size_t)((kt * 8 + nt) * 32 + lane) * 16));
            bf16x16 b = __builtin_bit_cast(bf16x16, *bp);
            acc[nt] = __builtin_amdgcn_wmma_f32_16x16x32_bf16(
                false, a, false, b, (short)0, acc[nt], false, false);
        }
    }

    // D layout: VGPR r -> row = r + 8*(lane>=16), col = lane&15
    const int rbase = (lane >> 4) << 3;
    const int col   = lane & 15;
#pragma unroll
    for (int nt = 0; nt < 8; ++nt) {
#pragma unroll
        for (int r = 0; r < 8; ++r) {
            H[(size_t)(m0 + rbase + r) * 128 + nt * 16 + col] = acc[nt][r];
        }
    }
}

// ---------------------------------------------------------------------------
// out[i,:] = h[i,:] * dis[i]^2   (self-loop term; also initializes out)
// ---------------------------------------------------------------------------
__global__ void self_init(const float* __restrict__ h, const float* __restrict__ dis,
                          float* __restrict__ out, int n32) {
    int idx = blockIdx.x * blockDim.x + threadIdx.x;
    if (idx >= n32) return;
    int node = idx >> 5;
    int c4   = (idx & 31) << 2;
    float d  = dis[node];
    float s  = d * d;
    const float4 v = *(const float4*)(h + (size_t)node * 128 + c4);
    float4 o;
    o.x = v.x * s; o.y = v.y * s; o.z = v.z * s; o.w = v.w * s;
    *(float4*)(out + (size_t)node * 128 + c4) = o;
}

// ---------------------------------------------------------------------------
// Edge scatter: out[dst] += h[src] * dis[src]*dis[dst]
// one thread = (edge, 4 channels); a wave covers 128 channels of one edge.
// ---------------------------------------------------------------------------
__global__ void edge_scatter(const int* __restrict__ src, const int* __restrict__ dst,
                             const float* __restrict__ dis, const float* __restrict__ h,
                             float* __restrict__ out, int E) {
    int idx = blockIdx.x * blockDim.x + threadIdx.x;
    if (idx >= E * 32) return;
    int e  = idx >> 5;
    int c4 = (idx & 31) << 2;
    int s  = src[e];
    int dd = dst[e];
    float nrm = dis[s] * dis[dd];
    const float4 v = *(const float4*)(h + (size_t)s * 128 + c4);
    float* op = out + (size_t)dd * 128 + c4;
    unsafeAtomicAdd(op + 0, v.x * nrm);
    unsafeAtomicAdd(op + 1, v.y * nrm);
    unsafeAtomicAdd(op + 2, v.z * nrm);
    unsafeAtomicAdd(op + 3, v.w * nrm);
}

// ---------------------------------------------------------------------------
// out = out + bias (+ ReLU); optionally also emit bf16 copy for next GEMM
// ---------------------------------------------------------------------------
__global__ void bias_act(float* __restrict__ out, const float* __restrict__ bias,
                         unsigned short* __restrict__ outb, int n32, int relu) {
    int idx = blockIdx.x * blockDim.x + threadIdx.x;
    if (idx >= n32) return;
    int node = idx >> 5;
    int c4   = (idx & 31) << 2;
    float4 v = *(float4*)(out + (size_t)node * 128 + c4);
    const float4 b = *(const float4*)(bias + c4);
    v.x += b.x; v.y += b.y; v.z += b.z; v.w += b.w;
    if (relu) {
        v.x = fmaxf(v.x, 0.0f); v.y = fmaxf(v.y, 0.0f);
        v.z = fmaxf(v.z, 0.0f); v.w = fmaxf(v.w, 0.0f);
    }
    *(float4*)(out + (size_t)node * 128 + c4) = v;
    if (outb) {
        u16x4 o;
        o[0] = f32_to_bf16(v.x); o[1] = f32_to_bf16(v.y);
        o[2] = f32_to_bf16(v.z); o[3] = f32_to_bf16(v.w);
        *(u16x4*)(outb + (size_t)node * 128 + c4) = o;
    }
}

// ---------------------------------------------------------------------------
// Launch
// ---------------------------------------------------------------------------
extern "C" void kernel_launch(void* const* d_in, const int* in_sizes, int n_in,
                              void* d_out, int out_size, void* d_ws, size_t ws_size,
                              hipStream_t stream) {
    const float* x  = (const float*)d_in[0];
    const int*   ei = (const int*)d_in[1];
    const float* W[3] = {(const float*)d_in[2], (const float*)d_in[4], (const float*)d_in[6]};
    const float* B[3] = {(const float*)d_in[3], (const float*)d_in[5], (const float*)d_in[7]};

    const int N = in_sizes[0] / 128;   // 100000
    const int E = in_sizes[1] / 2;     // 1600000

    // Workspace layout: dis[N] | Wf[3*16384 bf16] | h[N*128 f32] | xb[N*128 bf16]
    char* ws = (char*)d_ws;
    float* dis = (float*)ws;
    size_t off = (((size_t)N * 4) + 255) & ~(size_t)255;
    unsigned short* wf = (unsigned short*)(ws + off);
    off += (((size_t)3 * 16384 * 2) + 255) & ~(size_t)255;
    float* h = (float*)(ws + off);
    off += (((size_t)N * 128 * 4) + 255) & ~(size_t)255;
    unsigned short* xb = (unsigned short*)(ws + off);
    float* out = (float*)d_out;

    const int n32 = N * 32;

    // Normalization: dis = rsqrt(1 + in-degree)
    fill_ones  <<<(N + 255) / 256, 256, 0, stream>>>(dis, N);
    deg_scatter<<<(E + 255) / 256, 256, 0, stream>>>(ei + E, dis, E);
    deg_rsqrt  <<<(N + 255) / 256, 256, 0, stream>>>(dis, N);

    // Pre-swizzle the three weight matrices into WMMA B-fragment layout
    for (int l = 0; l < 3; ++l)
        swizzle_w<<<64, 256, 0, stream>>>(W[l], wf + (size_t)l * 16384);

    // Input activations -> bf16 once
    to_bf16<<<(n32 + 255) / 256, 256, 0, stream>>>(x, xb, n32);

    const int ntiles = N / 16;         // 6250, exact
    for (int l = 0; l < 3; ++l) {
        gemm_wmma_bf16<<<(ntiles + 7) / 8, 256, 0, stream>>>(xb, wf + (size_t)l * 16384, h, ntiles);
        self_init   <<<(n32 + 255) / 256, 256, 0, stream>>>(h, dis, out, n32);
        edge_scatter<<<(E * 32 + 255) / 256, 256, 0, stream>>>(ei, ei + E, dis, h, out, E);
        // layers 0,1: +bias, ReLU, and bf16 copy into xb for the next GEMM
        bias_act    <<<(n32 + 255) / 256, 256, 0, stream>>>(
            out, B[l], (l < 2) ? xb : (unsigned short*)nullptr, n32, (l < 2) ? 1 : 0);
    }
}
